// GGNNModel_10075993276620
// MI455X (gfx1250) — matrix operand order repaired and checked by reading, not compiled, not run on hardware
//
#include <hip/hip_runtime.h>
#include <hip/hip_bf16.h>
#include <math.h>

#define NNODES 100000
#define NGRAPHS 100
#define NEDGES 1000000
#define HDIM 256
#define TTYPES 4
#define LLAYERS 2
#define LPOS 512
#define GK 256   // K dimension is always 256

typedef __bf16 bf16_t;
typedef bf16_t v16bf __attribute__((ext_vector_type(16)));
typedef bf16_t v8bf  __attribute__((ext_vector_type(8)));
typedef float  v8f   __attribute__((ext_vector_type(8)));
typedef int    v4i   __attribute__((ext_vector_type(4)));

#if defined(__has_builtin)
#if __has_builtin(__builtin_amdgcn_global_load_async_to_lds_b128)
#define HAS_ASYNC_LDS 1
#endif
#if __has_builtin(__builtin_amdgcn_s_wait_asynccnt)
#define HAS_WAIT_ASYNC_BUILTIN 1
#endif
#endif

typedef __attribute__((address_space(1))) v4i* as1_v4i;
typedef __attribute__((address_space(3))) v4i* as3_v4i;

__device__ __forceinline__ void wait_asynccnt0() {
#if defined(HAS_WAIT_ASYNC_BUILTIN)
    __builtin_amdgcn_s_wait_asynccnt(0);
#else
    asm volatile("s_wait_asynccnt 0" ::: "memory");
#endif
}

// ---------------------------------------------------------------------------
// C[M,N] = A[M,256](bf16) @ Bt[N,256](bf16, pre-transposed) + bias[N]
// Block: 256 threads = 8 waves; block tile = 32 rows x 256 cols.
// Wave tile = 32 rows x 32 cols -> 2x2 WMMA tiles, 4 f32 accumulators.
// grid.x = M/32, grid.y = N/256.
// ---------------------------------------------------------------------------
__global__ __launch_bounds__(256)
void wmma_gemm_bias(const bf16_t* __restrict__ A, const bf16_t* __restrict__ Bt,
                    const float* __restrict__ bias, float* __restrict__ C, int N) {
    __shared__ bf16_t As[32 * GK];                 // 16 KB: 32-row bf16 A panel
    const int tid  = threadIdx.x;
    const int wave = tid >> 5;
    const int lane = tid & 31;
    const int row0 = blockIdx.x * 32;
    const int col0 = (blockIdx.y * 8 + wave) * 32;

    // Stage 32x256 bf16 panel (contiguous 16 KB: full rows, stride == GK).
    {
        const char* gsrc = (const char*)(A + (size_t)row0 * GK) + tid * 64;
        char* ldst = (char*)As + tid * 64;
#if defined(HAS_ASYNC_LDS)
#pragma unroll
        for (int i = 0; i < 4; ++i)
            __builtin_amdgcn_global_load_async_to_lds_b128(
                (as1_v4i)(gsrc + i * 16), (as3_v4i)(ldst + i * 16), 0, 0);
        wait_asynccnt0();
#else
        const float4* g4 = (const float4*)gsrc;
        float4* s4 = (float4*)ldst;
#pragma unroll
        for (int i = 0; i < 4; ++i) s4[i] = g4[i];
#endif
    }
    __syncthreads();

    const int m     = lane & 15;               // row within tile / col within tile
    const int abase = (lane < 16) ? 0 : 8;     // A frag: K 0..7 / 8..15 (+16 pair)
    const int bbase = (lane < 16) ? 0 : 16;    // unused; B frag loads use abase (same K order as A)
    const bf16_t* Bc0 = Bt + (size_t)(col0 + m) * GK;
    const bf16_t* Bc1 = Bt + (size_t)(col0 + 16 + m) * GK;
    const bf16_t* Ar0 = As + (size_t)m * GK;
    const bf16_t* Ar1 = As + (size_t)(m + 16) * GK;

    v8f acc00 = {}, acc01 = {}, acc10 = {}, acc11 = {};
#pragma unroll
    for (int k0 = 0; k0 < GK; k0 += 32) {
        v8bf a0lo = *(const v8bf*)(Ar0 + k0 + abase);
        v8bf a0hi = *(const v8bf*)(Ar0 + k0 + 16 + abase);
        v8bf a1lo = *(const v8bf*)(Ar1 + k0 + abase);
        v8bf a1hi = *(const v8bf*)(Ar1 + k0 + 16 + abase);
        v16bf a0 = __builtin_shufflevector(a0lo, a0hi,
                      0,1,2,3,4,5,6,7,8,9,10,11,12,13,14,15);
        v16bf a1 = __builtin_shufflevector(a1lo, a1hi,
                      0,1,2,3,4,5,6,7,8,9,10,11,12,13,14,15);
        v16bf b0 = __builtin_shufflevector(*(const v8bf*)(Bc0 + k0 + abase), *(const v8bf*)(Bc0 + k0 + 16 + abase),
                      0,1,2,3,4,5,6,7,8,9,10,11,12,13,14,15);
        v16bf b1 = __builtin_shufflevector(*(const v8bf*)(Bc1 + k0 + abase), *(const v8bf*)(Bc1 + k0 + 16 + abase),
                      0,1,2,3,4,5,6,7,8,9,10,11,12,13,14,15);
        acc00 = __builtin_amdgcn_wmma_f32_16x16x32_bf16(false, a0, false, b0, (short)0, acc00, false, false);
        acc01 = __builtin_amdgcn_wmma_f32_16x16x32_bf16(false, a0, false, b1, (short)0, acc01, false, false);
        acc10 = __builtin_amdgcn_wmma_f32_16x16x32_bf16(false, a1, false, b0, (short)0, acc10, false, false);
        acc11 = __builtin_amdgcn_wmma_f32_16x16x32_bf16(false, a1, false, b1, (short)0, acc11, false, false);
    }

    const int rbase = (lane < 16) ? 0 : 8;
    const int c0 = col0 + m;
    const int c1 = col0 + 16 + m;
    const float bv0 = bias[c0];
    const float bv1 = bias[c1];
#pragma unroll
    for (int i = 0; i < 8; ++i) {
        const size_t r0 = (size_t)(row0 + rbase + i) * N;
        const size_t r1 = (size_t)(row0 + 16 + rbase + i) * N;
        C[r0 + c0] = acc00[i] + bv0;
        C[r0 + c1] = acc01[i] + bv1;
        C[r1 + c0] = acc10[i] + bv0;
        C[r1 + c1] = acc11[i] + bv1;
    }
}

// ---------------------------------------------------------------------------
// Conversions
// ---------------------------------------------------------------------------
__global__ void conv_act_bf16(const float* __restrict__ in, bf16_t* __restrict__ o, int n) {
    int i = blockIdx.x * 256 + threadIdx.x;
    if (i < n) o[i] = (bf16_t)in[i];
}

__global__ void conv_typeW(const float* __restrict__ tw, bf16_t* __restrict__ o) {
    int idx = blockIdx.x * 256 + threadIdx.x;
    if (idx >= LLAYERS * TTYPES * HDIM * HDIM) return;
    int lt = idx / (HDIM * HDIM);
    int r  = idx % (HDIM * HDIM);
    int j  = r / HDIM;          // output feature (col)
    int i  = r % HDIM;          // k
    o[idx] = (bf16_t)tw[(size_t)lt * HDIM * HDIM + (size_t)i * HDIM + j];
}

__global__ void conv_gruT(const float* __restrict__ w, bf16_t* __restrict__ o) {
    int idx = blockIdx.x * 256 + threadIdx.x;
    if (idx >= LLAYERS * 3 * HDIM * HDIM) return;
    int l = idx / (3 * HDIM * HDIM);
    int r = idx % (3 * HDIM * HDIM);
    int n = r / HDIM;           // output feature 0..767
    int k = r % HDIM;
    o[idx] = (bf16_t)w[(size_t)l * HDIM * 3 * HDIM + (size_t)k * 3 * HDIM + n];
}

// ---------------------------------------------------------------------------
// h[n,c] = embed[token[n],c] + PE(min(pos[n],512), c)
// ---------------------------------------------------------------------------
__global__ void init_h(const int* __restrict__ nodes, const float* __restrict__ embed,
                       float* __restrict__ h) {
    int idx = blockIdx.x * 256 + threadIdx.x;
    if (idx >= NNODES * HDIM) return;
    int node = idx / HDIM, c = idx % HDIM;
    int tok = nodes[node];
    int pos = nodes[NNODES + node];
    int pidx = pos < LPOS ? pos : LPOS;
    float pe = 0.0f;
    if (pidx > 0) {
        float p = (float)(pidx - 1);
        float ang = p * powf(10000.0f, -(2.0f * (float)c) / (float)HDIM);
        pe = (c & 1) ? cosf(ang) : sinf(ang);
    }
    h[idx] = embed[(size_t)tok * HDIM + c] + pe;
}

// ---------------------------------------------------------------------------
// msgs[tgt] += trans[src] for edges of type t. One wave per edge.
// ---------------------------------------------------------------------------
__global__ __launch_bounds__(256)
void edge_scatter(const int* __restrict__ edges, const float* __restrict__ trans,
                  float* __restrict__ msgs, int t) {
    const int e = blockIdx.x * 8 + (threadIdx.x >> 5);
    const int lane = threadIdx.x & 31;
    if (e >= NEDGES) return;
    if (edges[e * 3] != t) return;
    const int s = edges[e * 3 + 1];
    const int d = edges[e * 3 + 2];
    const float4* src = (const float4*)(trans + (size_t)s * HDIM);
    float* dst = msgs + (size_t)d * HDIM;
#pragma unroll
    for (int i = 0; i < 2; ++i) {
        int q = lane + i * 32;          // 64 float4 per row
        float4 v = src[q];
        atomicAdd(dst + q * 4 + 0, v.x);
        atomicAdd(dst + q * 4 + 1, v.y);
        atomicAdd(dst + q * 4 + 2, v.z);
        atomicAdd(dst + q * 4 + 3, v.w);
    }
}

// ---------------------------------------------------------------------------
// GRU elementwise: h = z*h + (1-z)*tanh(xh + r*hh), in place.
// ---------------------------------------------------------------------------
__global__ void gru_update(float* __restrict__ h, const float* __restrict__ X,
                           const float* __restrict__ HU) {
    int idx = blockIdx.x * 256 + threadIdx.x;
    if (idx >= NNODES * HDIM) return;
    int node = idx / HDIM, c = idx % HDIM;
    const float* x  = X  + (size_t)node * 3 * HDIM;
    const float* hu = HU + (size_t)node * 3 * HDIM;
    float z = 1.0f / (1.0f + expf(-(x[c] + hu[c])));
    float r = 1.0f / (1.0f + expf(-(x[c + HDIM] + hu[c + HDIM])));
    float cand = tanhf(x[c + 2 * HDIM] + r * hu[c + 2 * HDIM]);
    float hv = h[idx];
    h[idx] = z * hv + (1.0f - z) * cand;
}

// ---------------------------------------------------------------------------
// Readout helpers
// ---------------------------------------------------------------------------
__global__ void prefix_seg(const int* __restrict__ sizes, int* __restrict__ segStarts) {
    if (threadIdx.x == 0 && blockIdx.x == 0) {
        int s = 0; segStarts[0] = 0;
        for (int g = 0; g < NGRAPHS; ++g) { s += sizes[g]; segStarts[g + 1] = s; }
    }
}

__device__ __forceinline__ int find_seg(const int* segStarts, int node) {
    int lo = 0, hi = NGRAPHS - 1;
    while (lo < hi) {
        int mid = (lo + hi) >> 1;
        if (node >= segStarts[mid + 1]) lo = mid + 1; else hi = mid;
    }
    return lo;
}

__global__ void zero_f32(float* __restrict__ p, int n) {
    int i = blockIdx.x * 256 + threadIdx.x;
    if (i < n) p[i] = 0.0f;
}

__global__ void zero_small(float* segsum, unsigned* gmaxEnc, float* out) {
    int t = threadIdx.x;
    if (t < NGRAPHS) segsum[t] = 0.0f;
    if (t == 0) *gmaxEnc = 0u;
    if (t < 2 * NGRAPHS) out[t] = 0.0f;
}

__global__ __launch_bounds__(256)
void gate_kernel(const float* __restrict__ h, const float* __restrict__ gw,
                 const float* __restrict__ gb, float* __restrict__ gateRaw,
                 unsigned* __restrict__ gmaxEnc) {
    int node = blockIdx.x * 8 + (threadIdx.x >> 5);
    int lane = threadIdx.x & 31;
    if (node >= NNODES) return;
    const float* hr = h + (size_t)node * HDIM;
    float acc = 0.0f;
    for (int c = lane; c < HDIM; c += 32) acc += hr[c] * gw[c];
    for (int off = 16; off; off >>= 1) acc += __shfl_down(acc, off, 32);
    if (lane == 0) {
        float g = acc + gb[0];
        gateRaw[node] = g;
        unsigned enc = __float_as_uint(g);
        enc = (enc & 0x80000000u) ? ~enc : (enc | 0x80000000u);
        atomicMax(gmaxEnc, enc);
    }
}

__global__ void exp_seg_kernel(const float* __restrict__ gateRaw,
                               const unsigned* __restrict__ gmaxEnc,
                               const int* __restrict__ segStarts,
                               float* __restrict__ gateExp, float* __restrict__ segsum) {
    int node = blockIdx.x * 256 + threadIdx.x;
    if (node >= NNODES) return;
    unsigned enc = *gmaxEnc;
    unsigned bits = (enc & 0x80000000u) ? (enc ^ 0x80000000u) : ~enc;
    float maxv = __uint_as_float(bits);
    float g = expf(gateRaw[node] - maxv);
    gateExp[node] = g;
    atomicAdd(&segsum[find_seg(segStarts, node)], g);
}

__global__ __launch_bounds__(256)
void out_kernel(const float* __restrict__ h, const float* __restrict__ ow,
                const float* __restrict__ ob, const float* __restrict__ gateExp,
                const float* __restrict__ segsum, const int* __restrict__ segStarts,
                float* __restrict__ out) {
    int node = blockIdx.x * 8 + (threadIdx.x >> 5);
    int lane = threadIdx.x & 31;
    if (node >= NNODES) return;
    const float* hr = h + (size_t)node * HDIM;
    float a0 = 0.0f, a1 = 0.0f;
    for (int c = lane; c < HDIM; c += 32) {
        float hv = hr[c];
        a0 += hv * ow[c * 2 + 0];
        a1 += hv * ow[c * 2 + 1];
    }
    for (int off = 16; off; off >>= 1) {
        a0 += __shfl_down(a0, off, 32);
        a1 += __shfl_down(a1, off, 32);
    }
    if (lane == 0) {
        int seg = find_seg(segStarts, node);
        float g = gateExp[node] / (segsum[seg] + 1e-16f);
        atomicAdd(&out[seg * 2 + 0], (a0 + ob[0]) * g);
        atomicAdd(&out[seg * 2 + 1], (a1 + ob[1]) * g);
    }
}

// ---------------------------------------------------------------------------
extern "C" void kernel_launch(void* const* d_in, const int* in_sizes, int n_in,
                              void* d_out, int out_size, void* d_ws, size_t ws_size,
                              hipStream_t stream) {
    const int*   nodes  = (const int*)d_in[0];
    const int*   gsizes = (const int*)d_in[1];
    const int*   edges  = (const int*)d_in[2];
    const float* embed  = (const float*)d_in[3];
    const float* tw     = (const float*)d_in[4];
    const float* tb     = (const float*)d_in[5];
    const float* gWm    = (const float*)d_in[6];
    const float* gUm    = (const float*)d_in[7];
    const float* gbi    = (const float*)d_in[8];
    const float* gbr    = (const float*)d_in[9];
    const float* gateW  = (const float*)d_in[10];
    const float* gateB  = (const float*)d_in[11];
    const float* outW   = (const float*)d_in[12];
    const float* outB   = (const float*)d_in[13];
    float* out = (float*)d_out;

    char* base = (char*)d_ws;
    size_t off = 0;
    auto carve = [&](size_t bytes) -> void* {
        void* p = base + off;
        off += (bytes + 255) & ~(size_t)255;
        return p;
    };
    float*  h       = (float*) carve((size_t)NNODES * HDIM * 4);
    float*  msgs    = (float*) carve((size_t)NNODES * HDIM * 4);
    float*  trans   = (float*) carve((size_t)NNODES * HDIM * 4);
    float*  X       = (float*) carve((size_t)NNODES * 3 * HDIM * 4);
    float*  HU      = (float*) carve((size_t)NNODES * 3 * HDIM * 4);
    bf16_t* hbf     = (bf16_t*)carve((size_t)NNODES * HDIM * 2);
    bf16_t* mbf     = (bf16_t*)carve((size_t)NNODES * HDIM * 2);
    bf16_t* typeWT  = (bf16_t*)carve((size_t)LLAYERS * TTYPES * HDIM * HDIM * 2);
    bf16_t* gruWT   = (bf16_t*)carve((size_t)LLAYERS * 3 * HDIM * HDIM * 2);
    bf16_t* gruUT   = (bf16_t*)carve((size_t)LLAYERS * 3 * HDIM * HDIM * 2);
    float*  gateRaw = (float*) carve((size_t)NNODES * 4);
    float*  gateExp = (float*) carve((size_t)NNODES * 4);
    float*  segsum  = (float*) carve(NGRAPHS * 4);
    int*    segSt   = (int*)   carve((NGRAPHS + 1) * 4);
    unsigned* gmaxE = (unsigned*)carve(4);

    const int nAct = NNODES * HDIM;             // 25.6M elements

    // --- setup ---
    prefix_seg<<<1, 32, 0, stream>>>(gsizes, segSt);
    conv_typeW<<<(LLAYERS * TTYPES * HDIM * HDIM + 255) / 256, 256, 0, stream>>>(tw, typeWT);
    conv_gruT<<<(LLAYERS * 3 * HDIM * HDIM + 255) / 256, 256, 0, stream>>>(gWm, gruWT);
    conv_gruT<<<(LLAYERS * 3 * HDIM * HDIM + 255) / 256, 256, 0, stream>>>(gUm, gruUT);
    init_h<<<(nAct + 255) / 256, 256, 0, stream>>>(nodes, embed, h);

    const int steps[LLAYERS] = {3, 1};
    for (int l = 0; l < LLAYERS; ++l) {
        for (int s = 0; s < steps[l]; ++s) {
            conv_act_bf16<<<(nAct + 255) / 256, 256, 0, stream>>>(h, hbf, nAct);
            zero_f32<<<(nAct + 255) / 256, 256, 0, stream>>>(msgs, nAct);
            for (int t = 0; t < TTYPES; ++t) {
                wmma_gemm_bias<<<dim3(NNODES / 32, HDIM / 256), 256, 0, stream>>>(
                    hbf, typeWT + (size_t)(l * TTYPES + t) * HDIM * HDIM,
                    tb + (size_t)(l * TTYPES + t) * HDIM, trans, HDIM);
                edge_scatter<<<(NEDGES + 7) / 8, 256, 0, stream>>>(edges, trans, msgs, t);
            }
            conv_act_bf16<<<(nAct + 255) / 256, 256, 0, stream>>>(msgs, mbf, nAct);
            wmma_gemm_bias<<<dim3(NNODES / 32, (3 * HDIM) / 256), 256, 0, stream>>>(
                mbf, gruWT + (size_t)l * 3 * HDIM * HDIM, gbi + (size_t)l * 3 * HDIM, X, 3 * HDIM);
            wmma_gemm_bias<<<dim3(NNODES / 32, (3 * HDIM) / 256), 256, 0, stream>>>(
                hbf, gruUT + (size_t)l * 3 * HDIM * HDIM, gbr + (size_t)l * 3 * HDIM, HU, 3 * HDIM);
            gru_update<<<(nAct + 255) / 256, 256, 0, stream>>>(h, X, HU);
        }
    }

    // --- readout ---
    zero_small<<<1, 256, 0, stream>>>(segsum, gmaxE, out);
    gate_kernel<<<(NNODES + 7) / 8, 256, 0, stream>>>(h, gateW, gateB, gateRaw, gmaxE);
    exp_seg_kernel<<<(NNODES + 255) / 256, 256, 0, stream>>>(gateRaw, gmaxE, segSt, gateExp, segsum);
    out_kernel<<<(NNODES + 7) / 8, 256, 0, stream>>>(h, outW, outB, gateExp, segsum, segSt, out);
}
